// SoftAggONNX_29678224016003
// MI455X (gfx1250) — matrix operand order repaired and checked by reading, not compiled, not run on hardware
//
#include <hip/hip_runtime.h>
#include <hip/hip_bf16.h>

#define NROWS 131072
#define DIM   512
#define G     4096
#define EPS   1e-12f
#define LSTR  520   // padded LDS row stride in bf16 elements (512 + 8)

// sched_group_barrier masks
#define SGB_WMMA    0x008
#define SGB_DSREAD  0x100

typedef __bf16 bf16_t;
typedef __attribute__((ext_vector_type(2)))  __bf16 v2bf;
typedef __attribute__((ext_vector_type(8)))  __bf16 v8bf;
typedef __attribute__((ext_vector_type(16))) __bf16 v16bf;
typedef __attribute__((ext_vector_type(8)))  float  v8f;

// ---- WMMA fragment loads from LDS (bf16, padded row stride LSTR) ----
// A tile 16x32: lanes 0-15 rows M=0..15 hold K = k0+0..15,
// lanes 16-31 same rows hold K = k0+16..31.
__device__ __forceinline__ v16bf load_frag_a(const bf16_t* base, int row, int kofs) {
    const bf16_t* p = base + row * LSTR + kofs + (kofs & 8);
    v8bf lo = *(const v8bf*)(p);
    v8bf hi = *(const v8bf*)(p + 8);
    return __builtin_shufflevector(lo, hi, 0,1,2,3,4,5,6,7,8,9,10,11,12,13,14,15);
}
// B tile 32x16 (stored as W rows = columns of B): lane l holds column n=l&15,
// K = k0..k0+15 for lanes 0-15, k0+16..k0+31 for lanes 16-31 (kofs pre-offset).
__device__ __forceinline__ v16bf load_frag_b(const bf16_t* base, int col, int kofs) {
    const bf16_t* p = base + col * LSTR + kofs;
    v8bf lo = *(const v8bf*)(p);
    v8bf hi = *(const v8bf*)(p + 8);
    return __builtin_shufflevector(lo, hi, 0,1,2,3,4,5,6,7,8,9,10,11,12,13,14,15);
}

__device__ __forceinline__ v8f wmma_bf16(v16bf a, v16bf b, v8f c) {
    return __builtin_amdgcn_wmma_f32_16x16x32_bf16(false, a, false, b, (short)0, c,
                                                   false, false);
}

// ---------------- Kernel 0: init group accumulators ----------------
__global__ __launch_bounds__(256) void k_init(float* gmax, float* denom, float* num) {
    int i = blockIdx.x * 256 + threadIdx.x;
    ((unsigned int*)gmax)[i] = 0xFF800000u;  // -inf (works for int-max / uint-min trick)
    denom[i] = 0.0f;
    num[i]   = 0.0f;
}

// ---------------- Kernel 1: logits GEMM + segment max ----------------
__global__ __launch_bounds__(256) void k_gemm_max(const float* __restrict__ x,
                                                  const float* __restrict__ Wg,
                                                  const float* __restrict__ bg,
                                                  const int*   __restrict__ jx,
                                                  float*       __restrict__ gmax) {
    __shared__ bf16_t xs[128 * LSTR];   // 130 KB: x slab as bf16
    __shared__ bf16_t wsl[64 * LSTR];   //  65 KB: weight slab (64 cols x 512 k)
    const int tid = threadIdx.x;
    const long rowBase = (long)blockIdx.x * 128;

    // stage x slab: fp32 -> bf16
    for (int p = tid; p < 128 * 256; p += 256) {
        int r = p >> 8, k2 = (p & 255) * 2;
        float2 v = *(const float2*)(x + (rowBase + r) * DIM + k2);
        v2bf h; h[0] = (bf16_t)v.x; h[1] = (bf16_t)v.y;
        *(v2bf*)(xs + r * LSTR + k2) = h;
    }

    const int lane = tid & 31, wave = tid >> 5;
    const int m0 = wave * 16;
    const int arow  = m0 + (lane & 15);
    const int aksel = (lane >> 4) << 3;
    const int bcol  = lane & 15;
    const int bksel = (lane >> 4) << 4;
    const int mhi   = (lane >> 4) << 3;

    for (int ng = 0; ng < 8; ++ng) {
        __syncthreads();
        for (int p = tid; p < 64 * 256; p += 256) {     // stage Wg[ng*64.., :]
            int r = p >> 8, k2 = (p & 255) * 2;
            float2 v = *(const float2*)(Wg + (long)(ng * 64 + r) * DIM + k2);
            v2bf h; h[0] = (bf16_t)v.x; h[1] = (bf16_t)v.y;
            *(v2bf*)(wsl + r * LSTR + k2) = h;
        }
        // prefetch next slab into L2 while we compute (global_prefetch_b8)
        {
            int ngn = (ng + 1) & 7;
            const float* nxt = Wg + (long)(ngn * 64) * DIM;
            __builtin_prefetch(nxt + (tid * 2 + 0) * 64, 0, 1);
            __builtin_prefetch(nxt + (tid * 2 + 1) * 64, 0, 1);
        }
        __syncthreads();

        v8f acc[4] = {};
        // software pipeline: fragments for step ks are loaded during step ks-32
        v16bf a_cur = load_frag_a(xs, arow, aksel);
        v16bf b_cur[4];
#pragma unroll
        for (int nt = 0; nt < 4; ++nt)
            b_cur[nt] = load_frag_b(wsl, nt * 16 + bcol, bksel);
#pragma unroll
        for (int ks = 0; ks < DIM; ks += 32) {
            const int ks2 = (ks + 32) & (DIM - 1);   // wraps on last iter (harmless)
            v16bf a_nxt = load_frag_a(xs, arow, ks2 + aksel);
            v16bf b_nxt[4];
#pragma unroll
            for (int nt = 0; nt < 4; ++nt)
                b_nxt[nt] = load_frag_b(wsl, nt * 16 + bcol, ks2 + bksel);
#pragma unroll
            for (int nt = 0; nt < 4; ++nt)
                acc[nt] = wmma_bf16(a_cur, b_cur[nt], acc[nt]);
            a_cur = a_nxt;
#pragma unroll
            for (int nt = 0; nt < 4; ++nt) b_cur[nt] = b_nxt[nt];
            // pin pipeline shape: next-step DS reads issue before this step's WMMAs
            __builtin_amdgcn_sched_group_barrier(SGB_DSREAD, 10, 0);
            __builtin_amdgcn_sched_group_barrier(SGB_WMMA,    4, 0);
        }

        float bias[4];
#pragma unroll
        for (int nt = 0; nt < 4; ++nt) bias[nt] = bg[ng * 64 + nt * 16 + bcol];
#pragma unroll
        for (int r = 0; r < 8; ++r) {
            long grow = rowBase + m0 + r + mhi;
            long jD = (long)jx[grow] * DIM;
#pragma unroll
            for (int nt = 0; nt < 4; ++nt) {
                int col = ng * 64 + nt * 16 + bcol;
                float v = acc[nt][r] + bias[nt];
                float* addr = gmax + jD + col;
                if (v >= 0.0f) atomicMax((int*)addr, __float_as_int(v));
                else           atomicMin((unsigned int*)addr, __float_as_uint(v));
            }
        }
    }
}

// ------- Kernel 2: recompute logits+feats, exp, segment-sum atomics -------
__global__ __launch_bounds__(256) void k_gemm_acc(const float* __restrict__ x,
                                                  const float* __restrict__ Wg,
                                                  const float* __restrict__ bg,
                                                  const float* __restrict__ Wf,
                                                  const float* __restrict__ bfv,
                                                  const int*   __restrict__ jx,
                                                  const float* __restrict__ gmax,
                                                  float*       __restrict__ denom,
                                                  float*       __restrict__ num) {
    __shared__ bf16_t xs[128 * LSTR];        // 130 KB
    __shared__ bf16_t wsl[2 * 64 * LSTR];    // 130 KB: [0]=Wg slab, [1]=Wf slab
    const int tid = threadIdx.x;
    const long rowBase = (long)blockIdx.x * 128;

    for (int p = tid; p < 128 * 256; p += 256) {
        int r = p >> 8, k2 = (p & 255) * 2;
        float2 v = *(const float2*)(x + (rowBase + r) * DIM + k2);
        v2bf h; h[0] = (bf16_t)v.x; h[1] = (bf16_t)v.y;
        *(v2bf*)(xs + r * LSTR + k2) = h;
    }

    const int lane = tid & 31, wave = tid >> 5;
    const int m0 = wave * 16;
    const int arow  = m0 + (lane & 15);
    const int aksel = (lane >> 4) << 3;
    const int bcol  = lane & 15;
    const int bksel = (lane >> 4) << 4;
    const int mhi   = (lane >> 4) << 3;

    for (int ng = 0; ng < 8; ++ng) {
        __syncthreads();
        for (int p = tid; p < 2 * 64 * 256; p += 256) {
            int mat = p >> 14;
            int rem = p & 16383;
            int r = rem >> 8, k2 = (rem & 255) * 2;
            const float* src = (mat == 0 ? Wg : Wf) + (long)(ng * 64 + r) * DIM + k2;
            float2 v = *(const float2*)src;
            v2bf h; h[0] = (bf16_t)v.x; h[1] = (bf16_t)v.y;
            *(v2bf*)(wsl + mat * 64 * LSTR + r * LSTR + k2) = h;
        }
        {
            int ngn = (ng + 1) & 7;
            __builtin_prefetch(Wg + (long)(ngn * 64) * DIM + (tid * 2 + 0) * 64, 0, 1);
            __builtin_prefetch(Wg + (long)(ngn * 64) * DIM + (tid * 2 + 1) * 64, 0, 1);
            __builtin_prefetch(Wf + (long)(ngn * 64) * DIM + (tid * 2 + 0) * 64, 0, 1);
            __builtin_prefetch(Wf + (long)(ngn * 64) * DIM + (tid * 2 + 1) * 64, 0, 1);
        }
        __syncthreads();

        v8f accG[4] = {}, accF[4] = {};
        v16bf a_cur = load_frag_a(xs, arow, aksel);
        v16bf bG_cur[4], bF_cur[4];
#pragma unroll
        for (int nt = 0; nt < 4; ++nt) {
            bG_cur[nt] = load_frag_b(wsl,             nt * 16 + bcol, bksel);
            bF_cur[nt] = load_frag_b(wsl + 64 * LSTR, nt * 16 + bcol, bksel);
        }
#pragma unroll
        for (int ks = 0; ks < DIM; ks += 32) {
            const int ks2 = (ks + 32) & (DIM - 1);
            v16bf a_nxt = load_frag_a(xs, arow, ks2 + aksel);
            v16bf bG_nxt[4], bF_nxt[4];
#pragma unroll
            for (int nt = 0; nt < 4; ++nt) {
                bG_nxt[nt] = load_frag_b(wsl,             nt * 16 + bcol, ks2 + bksel);
                bF_nxt[nt] = load_frag_b(wsl + 64 * LSTR, nt * 16 + bcol, ks2 + bksel);
            }
#pragma unroll
            for (int nt = 0; nt < 4; ++nt) {
                accG[nt] = wmma_bf16(a_cur, bG_cur[nt], accG[nt]);
                accF[nt] = wmma_bf16(a_cur, bF_cur[nt], accF[nt]);
            }
            a_cur = a_nxt;
#pragma unroll
            for (int nt = 0; nt < 4; ++nt) {
                bG_cur[nt] = bG_nxt[nt];
                bF_cur[nt] = bF_nxt[nt];
            }
            __builtin_amdgcn_sched_group_barrier(SGB_DSREAD, 18, 0);
            __builtin_amdgcn_sched_group_barrier(SGB_WMMA,    8, 0);
        }

        float biasG[4], biasF[4];
#pragma unroll
        for (int nt = 0; nt < 4; ++nt) {
            biasG[nt] = bg[ng * 64 + nt * 16 + bcol];
            biasF[nt] = bfv[ng * 64 + nt * 16 + bcol];
        }
#pragma unroll
        for (int r = 0; r < 8; ++r) {
            long grow = rowBase + m0 + r + mhi;
            long jD = (long)jx[grow] * DIM;
#pragma unroll
            for (int nt = 0; nt < 4; ++nt) {
                int col = ng * 64 + nt * 16 + bcol;
                float lg = accG[nt][r] + biasG[nt];
                float ft = accF[nt][r] + biasF[nt];
                float e = __expf(lg - gmax[jD + col]);
                atomicAdd(denom + jD + col, e);
                atomicAdd(num + jD + col, ft * e);
            }
        }
    }
}

// ------- Kernel 3: y = num/(denom+EPS); out_group = y @ Wh.T + bh -------
__global__ __launch_bounds__(256) void k_group_gemm(const float* __restrict__ num,
                                                    const float* __restrict__ denom,
                                                    const float* __restrict__ Wh,
                                                    const float* __restrict__ bh,
                                                    float*       __restrict__ outg) {
    __shared__ bf16_t ys[128 * LSTR];
    __shared__ bf16_t wsl[64 * LSTR];
    const int tid = threadIdx.x;
    const long rowBase = (long)blockIdx.x * 128;

    for (int p = tid; p < 128 * 256; p += 256) {
        int r = p >> 8, k2 = (p & 255) * 2;
        long idx = (rowBase + r) * DIM + k2;
        float2 nv = *(const float2*)(num + idx);
        float2 dv = *(const float2*)(denom + idx);
        v2bf h;
        h[0] = (bf16_t)(nv.x / (dv.x + EPS));
        h[1] = (bf16_t)(nv.y / (dv.y + EPS));
        *(v2bf*)(ys + r * LSTR + k2) = h;
    }

    const int lane = tid & 31, wave = tid >> 5;
    const int m0 = wave * 16;
    const int arow  = m0 + (lane & 15);
    const int aksel = (lane >> 4) << 3;
    const int bcol  = lane & 15;
    const int bksel = (lane >> 4) << 4;
    const int mhi   = (lane >> 4) << 3;

    for (int ng = 0; ng < 8; ++ng) {
        __syncthreads();
        for (int p = tid; p < 64 * 256; p += 256) {
            int r = p >> 8, k2 = (p & 255) * 2;
            float2 v = *(const float2*)(Wh + (long)(ng * 64 + r) * DIM + k2);
            v2bf h; h[0] = (bf16_t)v.x; h[1] = (bf16_t)v.y;
            *(v2bf*)(wsl + r * LSTR + k2) = h;
        }
        __syncthreads();

        v8f acc[4] = {};
        v16bf a_cur = load_frag_a(ys, arow, aksel);
        v16bf b_cur[4];
#pragma unroll
        for (int nt = 0; nt < 4; ++nt)
            b_cur[nt] = load_frag_b(wsl, nt * 16 + bcol, bksel);
#pragma unroll
        for (int ks = 0; ks < DIM; ks += 32) {
            const int ks2 = (ks + 32) & (DIM - 1);
            v16bf a_nxt = load_frag_a(ys, arow, ks2 + aksel);
            v16bf b_nxt[4];
#pragma unroll
            for (int nt = 0; nt < 4; ++nt)
                b_nxt[nt] = load_frag_b(wsl, nt * 16 + bcol, ks2 + bksel);
#pragma unroll
            for (int nt = 0; nt < 4; ++nt)
                acc[nt] = wmma_bf16(a_cur, b_cur[nt], acc[nt]);
            a_cur = a_nxt;
#pragma unroll
            for (int nt = 0; nt < 4; ++nt) b_cur[nt] = b_nxt[nt];
            __builtin_amdgcn_sched_group_barrier(SGB_DSREAD, 10, 0);
            __builtin_amdgcn_sched_group_barrier(SGB_WMMA,    4, 0);
        }
#pragma unroll
        for (int r = 0; r < 8; ++r) {
            long grow = rowBase + m0 + r + mhi;
#pragma unroll
            for (int nt = 0; nt < 4; ++nt) {
                int col = ng * 64 + nt * 16 + bcol;
                outg[grow * DIM + col] = acc[nt][r] + bh[col];
            }
        }
    }
}

// ---------------- Kernel 4: gather out[n,:] = out_group[jx[n],:] ----------------
__global__ __launch_bounds__(256) void k_gather(const float* __restrict__ outg,
                                                const int* __restrict__ jx,
                                                float* __restrict__ out) {
    long gid = (long)blockIdx.x * 256 + threadIdx.x;   // N*D/4 threads
    int n = (int)(gid >> 7);
    int c = (int)(gid & 127) << 2;
    long j = jx[n];
    float4 v = *(const float4*)(outg + j * DIM + c);
    *(float4*)(out + (long)n * DIM + c) = v;
}

extern "C" void kernel_launch(void* const* d_in, const int* in_sizes, int n_in,
                              void* d_out, int out_size, void* d_ws, size_t ws_size,
                              hipStream_t stream) {
    const float* x   = (const float*)d_in[0];
    const float* Wf  = (const float*)d_in[1];
    const float* bfv = (const float*)d_in[2];
    const float* Wg  = (const float*)d_in[3];
    const float* bg  = (const float*)d_in[4];
    const float* Wh  = (const float*)d_in[5];
    const float* bh  = (const float*)d_in[6];
    const int*   jx  = (const int*)d_in[7];
    float* out = (float*)d_out;

    float* gmax  = (float*)d_ws;                  // [G, D]
    float* denom = gmax  + (size_t)G * DIM;       // [G, D]
    float* num   = denom + (size_t)G * DIM;       // [G, D]
    float* outg  = num   + (size_t)G * DIM;       // [G, D]

    k_init<<<(G * DIM) / 256, 256, 0, stream>>>(gmax, denom, num);
    k_gemm_max<<<NROWS / 128, 256, 0, stream>>>(x, Wg, bg, jx, gmax);
    k_gemm_acc<<<NROWS / 128, 256, 0, stream>>>(x, Wg, bg, Wf, bfv, jx, gmax, denom, num);
    k_group_gemm<<<G / 128, 256, 0, stream>>>(num, denom, Wh, bh, outg);
    k_gather<<<(long)NROWS * DIM / 4 / 256, 256, 0, stream>>>(outg, jx, out);
}
